// IntentContradictionNet_59940563583141
// MI455X (gfx1250) — matrix operand order repaired and checked by reading, not compiled, not run on hardware
//
#include <hip/hip_runtime.h>
#include <hip/hip_bf16.h>
#include <math.h>

typedef __attribute__((ext_vector_type(16))) __bf16 v16bf;
typedef __attribute__((ext_vector_type(8)))  __bf16 v8bf;
typedef __attribute__((ext_vector_type(8)))  float  v8f;

// ---------------------------------------------------------------------------
// helpers
// ---------------------------------------------------------------------------
__device__ __forceinline__ float wred(float v) {
#pragma unroll
  for (int i = 16; i > 0; i >>= 1) v += __shfl_xor(v, i, 32);
  return v;
}

__device__ __forceinline__ float gelu_f(float x) {
  return 0.5f * x * (1.0f + erff(x * 0.70710678118654752440f));
}

// A fragment: 16x32 bf16, row-major A [M,K], lda in elements.
// lane l (l<16): row l, K = k0 + {h*8+0..7} (regs0-3) and k0+16+h*8+0..7 (regs4-7)
__device__ __forceinline__ v16bf load_a16(const __bf16* __restrict__ A, int lda,
                                          int m0, int k0, int lane) {
  int r = lane & 15, h = lane >> 4;
  const __bf16* p = A + (size_t)(m0 + r) * lda + k0 + h * 8;
  v8bf lo = *(const v8bf*)p;
  v8bf hi = *(const v8bf*)(p + 16);
  v16bf f;
#pragma unroll
  for (int i = 0; i < 8; ++i) { f[i] = lo[i]; f[i + 8] = hi[i]; }
  return f;
}

// B fragment: 32x16 bf16 from B^T stored [N,K] row-major (ldb elements).
// lane l: column n=l&15, K = k0 + h*8 + 0..7 (regs0-3) and k0+16+h*8+0..7 (regs4-7)
__device__ __forceinline__ v16bf load_bt16(const __bf16* __restrict__ Bt, int ldb,
                                           int n0, int k0, int lane) {
  int n = lane & 15, h = lane >> 4;
  const __bf16* p = Bt + (size_t)(n0 + n) * ldb + k0 + h * 8;
  v8bf lo = *(const v8bf*)p;
  v8bf hi = *(const v8bf*)(p + 16);
  v16bf f;
#pragma unroll
  for (int i = 0; i < 8; ++i) { f[i] = lo[i]; f[i + 8] = hi[i]; }
  return f;
}

// ---------------------------------------------------------------------------
// weight prep: fp32 -> bf16 repacks (one-shot, deterministic every call)
// ---------------------------------------------------------------------------
__global__ __launch_bounds__(256) void prep_kernel(
    const float* __restrict__ Wsh, const float* __restrict__ Wr1,
    const float* __restrict__ dW, const float* __restrict__ uW,
    const float* __restrict__ cw1, const float* __restrict__ cw3,
    const float* __restrict__ cw5, __bf16* __restrict__ Wshb,
    __bf16* __restrict__ Wr1pb, float* __restrict__ w0,
    __bf16* __restrict__ dWb, __bf16* __restrict__ uWb,
    float* __restrict__ convT) {
  int idx = blockIdx.x * 256 + threadIdx.x;
  if (idx < 65536) {
    Wshb[idx] = (__bf16)Wsh[idx];
    int n = idx >> 8, kk = idx & 255;
    Wr1pb[idx] = (__bf16)Wr1[n * 257 + 1 + kk];
    if (kk == 0) w0[n] = Wr1[n * 257];
  }
  if (idx < 49152) {
    dWb[idx] = (__bf16)dW[idx];
    uWb[idx] = (__bf16)uW[idx];
  }
  if (idx < 36864) {  // conv repack: [rout][rin][j] -> [j][rin][rout]
    int ksz; const float* src; int off;
    if (idx < 4096)       { ksz = 1; src = cw1; off = 0; }
    else if (idx < 16384) { ksz = 3; src = cw3; off = 4096; }
    else                  { ksz = 5; src = cw5; off = 16384; }
    int rem = idx - off;
    int rout = rem & 63, rin = (rem >> 6) & 63, j = rem >> 12;
    convT[idx] = src[(rout * 64 + rin) * ksz + j];
  }
}

// ---------------------------------------------------------------------------
// LayerNorm rows -> bf16 (wave per row)
// ---------------------------------------------------------------------------
__global__ __launch_bounds__(256) void ln_to_bf16(
    const float* __restrict__ X, const float* __restrict__ g,
    const float* __restrict__ b, __bf16* __restrict__ Y, int rows) {
  int row = blockIdx.x * 8 + (threadIdx.x >> 5);
  if (row >= rows) return;
  int lane = threadIdx.x & 31;
  const float* xr = X + (size_t)row * 256;
  float y[8], s = 0.f, sq = 0.f;
#pragma unroll
  for (int j = 0; j < 8; ++j) {
    float v = xr[lane + 32 * j];
    y[j] = v; s += v; sq += v * v;
  }
  s = wred(s); sq = wred(sq);
  float mean = s * (1.f / 256.f);
  float var  = sq * (1.f / 256.f) - mean * mean;
  float rstd = rsqrtf(var + 1e-5f);
  __bf16* yr = Y + (size_t)row * 256;
#pragma unroll
  for (int j = 0; j < 8; ++j) {
    int d = lane + 32 * j;
    yr[d] = (__bf16)((y[j] - mean) * rstd * g[d] + b[d]);
  }
}

// ---------------------------------------------------------------------------
// generic WMMA GEMM, 16(M) x 64(N) strip per wave:
//   C[M,N] = A[M,K] (bf16 rm) * Bt[N,K]^T (bf16 rm)
// A fragment loaded once per k-step, reused by 4 accumulators.
// EPI: 0 = +bias, 1 = (acc-1)*scale (logK), 2 = gelu(acc+bias+S[m]*w0[n]),
//      3 = gelu(acc+bias)
// ---------------------------------------------------------------------------
template <int EPI>
__global__ __launch_bounds__(32) void gemm_bt(
    const __bf16* __restrict__ A, const __bf16* __restrict__ Bt,
    float* __restrict__ C, int K, int lda, int ldb, int ldc,
    size_t sA, size_t sB, size_t sC,
    const float* __restrict__ bias, const float* __restrict__ rowvec,
    const float* __restrict__ w0col, float scale) {
  int lane = threadIdx.x;
  int m0 = blockIdx.x * 16, n0 = blockIdx.y * 64, bz = blockIdx.z;
  A += (size_t)bz * sA; Bt += (size_t)bz * sB; C += (size_t)bz * sC;
  v8f c[4];
#pragma unroll
  for (int ns = 0; ns < 4; ++ns)
#pragma unroll
    for (int i = 0; i < 8; ++i) c[ns][i] = 0.f;
  for (int k0 = 0; k0 < K; k0 += 32) {
    v16bf a = load_a16(A, lda, m0, k0, lane);
#pragma unroll
    for (int ns = 0; ns < 4; ++ns) {
      v16bf bb = load_bt16(Bt, ldb, n0 + ns * 16, k0, lane);
      c[ns] = __builtin_amdgcn_wmma_f32_16x16x32_bf16(false, a, false, bb,
                                                      (short)0, c[ns], false,
                                                      false);
    }
  }
  int mb = m0 + (lane >> 4) * 8;
#pragma unroll
  for (int ns = 0; ns < 4; ++ns) {
    int n = n0 + ns * 16 + (lane & 15);
#pragma unroll
    for (int i = 0; i < 8; ++i) {
      int m = mb + i;
      float v = c[ns][i];
      if (EPI == 0) v += bias[n];
      if (EPI == 1) v = (v - 1.0f) * scale;
      if (EPI == 2) v = gelu_f(v + bias[n] + rowvec[m] * w0col[n]);
      if (EPI == 3) v = gelu_f(v + bias[n]);
      C[(size_t)m * ldc + n] = v;
    }
  }
}

// ---------------------------------------------------------------------------
// row-normalize to bf16 (text); audio variant also writes Ap^T bf16 [B][D][La]
// ---------------------------------------------------------------------------
__global__ __launch_bounds__(256) void norm_rows(const float* __restrict__ X,
                                                 __bf16* __restrict__ Y, int rows) {
  int row = blockIdx.x * 8 + (threadIdx.x >> 5);
  if (row >= rows) return;
  int lane = threadIdx.x & 31;
  const float* xr = X + (size_t)row * 256;
  float y[8], sq = 0.f;
#pragma unroll
  for (int j = 0; j < 8; ++j) { float v = xr[lane + 32 * j]; y[j] = v; sq += v * v; }
  float nr = fmaxf(sqrtf(wred(sq)), 1e-12f);
  float inv = 1.f / nr;
  __bf16* yr = Y + (size_t)row * 256;
#pragma unroll
  for (int j = 0; j < 8; ++j) yr[lane + 32 * j] = (__bf16)(y[j] * inv);
}

__global__ __launch_bounds__(256) void norm_audio(const float* __restrict__ Ap,
                                                  __bf16* __restrict__ Anb,
                                                  __bf16* __restrict__ ApTb,
                                                  int rows) {
  int row = blockIdx.x * 8 + (threadIdx.x >> 5);
  if (row >= rows) return;
  int lane = threadIdx.x & 31;
  int b = row >> 11, n = row & 2047;
  const float* xr = Ap + (size_t)row * 256;
  float y[8], sq = 0.f;
#pragma unroll
  for (int j = 0; j < 8; ++j) { float v = xr[lane + 32 * j]; y[j] = v; sq += v * v; }
  float inv = 1.f / fmaxf(sqrtf(wred(sq)), 1e-12f);
  __bf16* yr = Anb + (size_t)row * 256;
#pragma unroll
  for (int j = 0; j < 8; ++j) {
    int d = lane + 32 * j;
    yr[d] = (__bf16)(y[j] * inv);
    ApTb[((size_t)b * 256 + d) * 2048 + n] = (__bf16)y[j];  // un-normalized Ap^T
  }
}

// ---------------------------------------------------------------------------
// Sinkhorn sweeps (logK fp32 lives in L2: 134MB < 192MB)
// ---------------------------------------------------------------------------
__global__ __launch_bounds__(256) void sink_row(const float* __restrict__ logK,
                                                const float* __restrict__ lv,
                                                float* __restrict__ lu,
                                                float log_mu) {
  __shared__ float sm[256], ss[256];
  int row = blockIdx.x;                 // b*1024 + m
  int b = row >> 10;
  const float* rk = logK + (size_t)row * 2048;
  const float* lvb = lv + b * 2048;
  int t = threadIdx.x;
  float mx = -3.4e38f, s = 0.f;
#pragma unroll
  for (int j = 0; j < 8; ++j) {
    float x = rk[t + 256 * j] + lvb[t + 256 * j];
    if (x > mx) { s = s * expf(mx - x) + 1.f; mx = x; }
    else        { s += expf(x - mx); }
  }
  sm[t] = mx; ss[t] = s;
  __syncthreads();
  for (int off = 128; off > 0; off >>= 1) {
    if (t < off) {
      float m1 = sm[t], s1 = ss[t], m2 = sm[t + off], s2 = ss[t + off];
      float M = fmaxf(m1, m2);
      ss[t] = s1 * expf(m1 - M) + s2 * expf(m2 - M);
      sm[t] = M;
    }
    __syncthreads();
  }
  if (t == 0) {
    float r = log_mu - (sm[0] + logf(ss[0]));
    r = fminf(fmaxf(r, -1e9f), 1e9f);
    if (r != r) r = 0.f;
    lu[row] = r;
  }
}

__global__ __launch_bounds__(256) void sink_col(const float* __restrict__ logK,
                                                const float* __restrict__ lu,
                                                float* __restrict__ lv,
                                                float log_nu) {
  __shared__ float lus[1024];
  int b = blockIdx.y;
  int n = blockIdx.x * 256 + threadIdx.x;
  for (int j = threadIdx.x; j < 1024; j += 256) lus[j] = lu[b * 1024 + j];
  __syncthreads();
  const float* base = logK + (size_t)b * 1024 * 2048 + n;
  float mx = -3.4e38f, s = 0.f;
  for (int m = 0; m < 1024; ++m) {
    float x = base[(size_t)m * 2048] + lus[m];
    if (x > mx) { s = s * expf(mx - x) + 1.f; mx = x; }
    else        { s += expf(x - mx); }
  }
  float r = log_nu - (mx + logf(s));
  r = fminf(fmaxf(r, -1e9f), 1e9f);
  if (r != r) r = 0.f;
  lv[b * 2048 + n] = r;
}

__global__ __launch_bounds__(256) void zero2(float* a, int na, float* b, int nb) {
  int i = blockIdx.x * 256 + threadIdx.x;
  if (i < na) a[i] = 0.f;
  if (i < nb) b[i] = 0.f;
}

// ---------------------------------------------------------------------------
// A_al = pi @ Ap : pi built on the fly = exp(lu+logK+lv) -> bf16 A-fragment
// wave computes a 16(M) x 64(N) strip (A fragment reused across 4 N tiles)
// ---------------------------------------------------------------------------
__global__ __launch_bounds__(32) void gemm_aal(const float* __restrict__ logK,
                                               const float* __restrict__ lu,
                                               const float* __restrict__ lv,
                                               const __bf16* __restrict__ ApT,
                                               float* __restrict__ Aal) {
  const int Lt = 1024, La = 2048, Dm = 256;
  int lane = threadIdx.x;
  int m0 = blockIdx.x * 16, n0 = blockIdx.y * 64, b = blockIdx.z;
  int r = lane & 15, h = lane >> 4;
  int m = m0 + r;
  const float* rowK = logK + ((size_t)b * Lt + m) * La;
  const float* lvb = lv + b * La;
  float lum = lu[b * Lt + m];
  const __bf16* Bt = ApT + (size_t)b * Dm * La;
  float* Cb = Aal + (size_t)b * Lt * Dm;
  v8f c[4];
#pragma unroll
  for (int ns = 0; ns < 4; ++ns)
#pragma unroll
    for (int i = 0; i < 8; ++i) c[ns][i] = 0.f;
  for (int k0 = 0; k0 < La; k0 += 32) {
    v16bf a;
    int kb0 = k0 + h * 8, kb1 = k0 + 16 + h * 8;
#pragma unroll
    for (int i = 0; i < 8; ++i) {
      float p0 = expf(rowK[kb0 + i] + lvb[kb0 + i] + lum);
      if (!(p0 < 3.0e38f)) p0 = (p0 != p0) ? 0.f : 1.0f;
      a[i] = (__bf16)p0;
      float p1 = expf(rowK[kb1 + i] + lvb[kb1 + i] + lum);
      if (!(p1 < 3.0e38f)) p1 = (p1 != p1) ? 0.f : 1.0f;
      a[i + 8] = (__bf16)p1;
    }
#pragma unroll
    for (int ns = 0; ns < 4; ++ns) {
      v16bf bb = load_bt16(Bt, La, n0 + ns * 16, k0, lane);
      c[ns] = __builtin_amdgcn_wmma_f32_16x16x32_bf16(false, a, false, bb,
                                                      (short)0, c[ns], false,
                                                      false);
    }
  }
  int mb = m0 + h * 8, nn = lane & 15;
#pragma unroll
  for (int ns = 0; ns < 4; ++ns)
#pragma unroll
    for (int i = 0; i < 8; ++i)
      Cb[(size_t)(mb + i) * Dm + n0 + ns * 16 + nn] = c[ns][i];
}

// ---------------------------------------------------------------------------
// S = sum(Tp*Aal), resid = |Tp - Aal| (f32 + bf16)
// ---------------------------------------------------------------------------
__global__ __launch_bounds__(256) void resid_S(const float* __restrict__ Tp,
                                               const float* __restrict__ Aal,
                                               float* __restrict__ S,
                                               float* __restrict__ resid,
                                               __bf16* __restrict__ residb,
                                               int rows) {
  int row = blockIdx.x * 8 + (threadIdx.x >> 5);
  if (row >= rows) return;
  int lane = threadIdx.x & 31;
  size_t base = (size_t)row * 256;
  float dot = 0.f;
#pragma unroll
  for (int j = 0; j < 8; ++j) {
    int d = lane + 32 * j;
    float t = Tp[base + d], a = Aal[base + d];
    dot += t * a;
    float rr = fabsf(t - a);
    resid[base + d] = rr;
    residb[base + d] = (__bf16)rr;
  }
  dot = wred(dot);
  if (lane == 0) S[row] = dot;
}

// ---------------------------------------------------------------------------
// router layer 2 + softmax (E=3), wave per row
// ---------------------------------------------------------------------------
__global__ __launch_bounds__(256) void router2(const float* __restrict__ H,
                                               const float* __restrict__ Wr2,
                                               const float* __restrict__ br2,
                                               float* __restrict__ G, int rows) {
  int row = blockIdx.x * 8 + (threadIdx.x >> 5);
  if (row >= rows) return;
  int lane = threadIdx.x & 31;
  size_t base = (size_t)row * 256;
  float p0 = 0.f, p1 = 0.f, p2 = 0.f;
#pragma unroll
  for (int j = 0; j < 8; ++j) {
    int d = lane + 32 * j;
    float h = H[base + d];
    p0 += h * Wr2[d];
    p1 += h * Wr2[256 + d];
    p2 += h * Wr2[512 + d];
  }
  p0 = wred(p0); p1 = wred(p1); p2 = wred(p2);
  if (lane == 0) {
    float l0 = p0 + br2[0], l1 = p1 + br2[1], l2 = p2 + br2[2];
    float mx = fmaxf(l0, fmaxf(l1, l2));
    float e0 = expf(l0 - mx), e1 = expf(l1 - mx), e2 = expf(l2 - mx);
    float inv = 1.f / (e0 + e1 + e2);
    G[row * 3 + 0] = e0 * inv;
    G[row * 3 + 1] = e1 * inv;
    G[row * 3 + 2] = e2 * inv;
  }
}

// ---------------------------------------------------------------------------
// expert conv: block = 64 rout threads for one (b,t); taps staged in LDS
// weights repacked [j][rin][rout] for broadcast/coalesced reads
// ---------------------------------------------------------------------------
__global__ __launch_bounds__(64) void conv_expert(const float* __restrict__ h,
                                                  const float* __restrict__ wT,
                                                  const float* __restrict__ cb,
                                                  __bf16* __restrict__ gbf,
                                                  int ksz, int pad) {
  __shared__ float hbuf[5 * 64];
  int t = blockIdx.x & 1023, b = blockIdx.x >> 10;
  int rid = threadIdx.x;
  for (int j = 0; j < ksz; ++j) {
    int tt = t - pad + j;
    hbuf[j * 64 + rid] =
        (tt >= 0 && tt < 1024) ? h[((size_t)(b * 1024 + tt)) * 64 + rid] : 0.f;
  }
  __syncthreads();
  float acc = cb[rid];
  for (int j = 0; j < ksz; ++j)
#pragma unroll
    for (int rin = 0; rin < 64; ++rin)
      acc += wT[(j * 64 + rin) * 64 + rid] * hbuf[j * 64 + rin];
  gbf[((size_t)blockIdx.x) * 64 + rid] = (__bf16)gelu_f(acc);
}

// ---------------------------------------------------------------------------
// expert epilogue: out (+)= G[:,e] * LN(hu + resid), wave per row
// ---------------------------------------------------------------------------
__global__ __launch_bounds__(256) void expert_out(const float* __restrict__ hu,
                                                  const float* __restrict__ resid,
                                                  const float* __restrict__ gl,
                                                  const float* __restrict__ bl,
                                                  const float* __restrict__ G,
                                                  float* __restrict__ out,
                                                  int e, int accum, int rows) {
  int row = blockIdx.x * 8 + (threadIdx.x >> 5);
  if (row >= rows) return;
  int lane = threadIdx.x & 31;
  size_t base = (size_t)row * 256;
  float y[8], s = 0.f, sq = 0.f;
#pragma unroll
  for (int j = 0; j < 8; ++j) {
    int d = lane + 32 * j;
    float v = hu[base + d] + resid[base + d];
    y[j] = v; s += v; sq += v * v;
  }
  s = wred(s); sq = wred(sq);
  float mean = s * (1.f / 256.f);
  float var  = sq * (1.f / 256.f) - mean * mean;
  float rstd = rsqrtf(var + 1e-5f);
  float gate = G[row * 3 + e];
#pragma unroll
  for (int j = 0; j < 8; ++j) {
    int d = lane + 32 * j;
    float eo = (y[j] - mean) * rstd * gl[d] + bl[d];
    size_t idx = base + d;
    if (accum) out[idx] += gate * eo;
    else       out[idx] = gate * eo;
  }
}

// ---------------------------------------------------------------------------
// host orchestration
// ---------------------------------------------------------------------------
extern "C" void kernel_launch(void* const* d_in, const int* in_sizes, int n_in,
                              void* d_out, int out_size, void* d_ws, size_t ws_size,
                              hipStream_t stream) {
  (void)in_sizes; (void)n_in; (void)out_size; (void)ws_size;
  const float* text  = (const float*)d_in[0];
  const float* audio = (const float*)d_in[1];
  const float* ln_g  = (const float*)d_in[2];
  const float* ln_b  = (const float*)d_in[3];
  const float* W_sh  = (const float*)d_in[4];
  const float* b_sh  = (const float*)d_in[5];
  const float* Wr1   = (const float*)d_in[6];
  const float* br1   = (const float*)d_in[7];
  const float* Wr2   = (const float*)d_in[8];
  const float* br2   = (const float*)d_in[9];
  const float* down_W = (const float*)d_in[10];
  const float* down_b = (const float*)d_in[11];
  const float* cw1   = (const float*)d_in[12];
  const float* cw3   = (const float*)d_in[13];
  const float* cw5   = (const float*)d_in[14];
  const float* conv_b = (const float*)d_in[15];
  const float* up_W  = (const float*)d_in[16];
  const float* up_b  = (const float*)d_in[17];
  const float* en_g  = (const float*)d_in[18];
  const float* en_b  = (const float*)d_in[19];
  float* out = (float*)d_out;

  constexpr int B = 16, Lt = 1024, La = 2048, Dm = 256, R = 64;
  const int Mt = B * Lt, Ma = B * La;

  char* wp = (char*)d_ws;
  auto alloc = [&](size_t bytes) -> void* {
    void* r = (void*)wp;
    wp += (bytes + 255) & ~(size_t)255;
    return r;
  };
  float*  Tp    = (float*)alloc((size_t)Mt * Dm * 4);
  float*  Ap    = (float*)alloc((size_t)Ma * Dm * 4);
  __bf16* Xln   = (__bf16*)alloc((size_t)Ma * Dm * 2);  // later aliased as Hr
  float*  Hr    = (float*)Xln;
  __bf16* Tnb   = (__bf16*)alloc((size_t)Mt * Dm * 2);
  __bf16* Anb   = (__bf16*)alloc((size_t)Ma * Dm * 2);  // later aliased as hu
  float*  hu    = (float*)Anb;
  __bf16* ApTb  = (__bf16*)alloc((size_t)B * Dm * La * 2);
  float*  logK  = (float*)alloc((size_t)B * Lt * La * 4);
  float*  lu    = (float*)alloc((size_t)Mt * 4);
  float*  lv    = (float*)alloc((size_t)Ma * 4);
  float*  Aal   = (float*)alloc((size_t)Mt * Dm * 4);
  float*  S     = (float*)alloc((size_t)Mt * 4);
  float*  resid = (float*)alloc((size_t)Mt * Dm * 4);
  __bf16* residb = (__bf16*)alloc((size_t)Mt * Dm * 2);
  float*  G     = (float*)alloc((size_t)Mt * 3 * 4);
  float*  hE    = (float*)alloc((size_t)Mt * R * 4);
  __bf16* gbf   = (__bf16*)alloc((size_t)Mt * R * 2);
  __bf16* Wshb  = (__bf16*)alloc(65536 * 2);
  __bf16* Wr1pb = (__bf16*)alloc(65536 * 2);
  float*  w0    = (float*)alloc(256 * 4);
  __bf16* dWb   = (__bf16*)alloc(49152 * 2);
  __bf16* uWb   = (__bf16*)alloc(49152 * 2);
  float*  convT = (float*)alloc(36864 * 4);

  prep_kernel<<<256, 256, 0, stream>>>(W_sh, Wr1, down_W, up_W, cw1, cw3, cw5,
                                       Wshb, Wr1pb, w0, dWb, uWb, convT);

  // shared projections
  ln_to_bf16<<<Mt / 8, 256, 0, stream>>>(text, ln_g, ln_b, Xln, Mt);
  gemm_bt<0><<<dim3(Mt / 16, Dm / 64, 1), 32, 0, stream>>>(
      Xln, Wshb, Tp, 256, 256, 256, 256, 0, 0, 0, b_sh, nullptr, nullptr, 0.f);
  norm_rows<<<Mt / 8, 256, 0, stream>>>(Tp, Tnb, Mt);

  ln_to_bf16<<<Ma / 8, 256, 0, stream>>>(audio, ln_g, ln_b, Xln, Ma);
  gemm_bt<0><<<dim3(Ma / 16, Dm / 64, 1), 32, 0, stream>>>(
      Xln, Wshb, Ap, 256, 256, 256, 256, 0, 0, 0, b_sh, nullptr, nullptr, 0.f);
  norm_audio<<<Ma / 8, 256, 0, stream>>>(Ap, Anb, ApTb, Ma);

  // logK = (cos - 1)/eps  (batched 1024x2048x256 bf16 WMMA GEMM)
  gemm_bt<1><<<dim3(Lt / 16, La / 64, B), 32, 0, stream>>>(
      Tnb, Anb, logK, 256, 256, 256, La, (size_t)Lt * 256, (size_t)La * 256,
      (size_t)Lt * La, nullptr, nullptr, nullptr, 1.0f / (0.1f + 1e-8f));

  // sinkhorn
  zero2<<<(Ma + 255) / 256, 256, 0, stream>>>(lu, Mt, lv, Ma);
  for (int it = 0; it < 10; ++it) {
    sink_row<<<Mt, 256, 0, stream>>>(logK, lv, lu, -6.93147180559945f);
    sink_col<<<dim3(La / 256, B), 256, 0, stream>>>(logK, lu, lv,
                                                    -7.62461898615940f);
  }

  // A_al = pi @ Ap (pi built on the fly)
  gemm_aal<<<dim3(Lt / 16, Dm / 64, B), 32, 0, stream>>>(logK, lu, lv, ApTb, Aal);
  resid_S<<<Mt / 8, 256, 0, stream>>>(Tp, Aal, S, resid, residb, Mt);

  // router  (Hr aliases Xln, dead by now)
  gemm_bt<2><<<dim3(Mt / 16, Dm / 64, 1), 32, 0, stream>>>(
      residb, Wr1pb, Hr, 256, 256, 256, 256, 0, 0, 0, br1, S, w0, 0.f);
  router2<<<Mt / 8, 256, 0, stream>>>(Hr, Wr2, br2, G, Mt);

  // experts  (hu aliases Anb, dead after logK GEMM)
  const int koff[3] = {0, 4096, 16384};
  const int ksz[3]  = {1, 3, 5};
  for (int e = 0; e < 3; ++e) {
    gemm_bt<3><<<dim3(Mt / 16, R / 64, 1), 32, 0, stream>>>(
        residb, dWb + e * R * 256, hE, 256, 256, 256, R, 0, 0, 0,
        down_b + e * R, nullptr, nullptr, 0.f);
    conv_expert<<<Mt, 64, 0, stream>>>(hE, convT + koff[e], conv_b + e * R, gbf,
                                       ksz[e], (ksz[e] - 1) / 2);
    gemm_bt<0><<<dim3(Mt / 16, Dm / 64, 1), 32, 0, stream>>>(
        gbf, uWb + (size_t)e * Dm * R, hu, R, R, R, Dm, 0, 0, 0, up_b + e * Dm,
        nullptr, nullptr, 0.f);
    expert_out<<<Mt / 8, 256, 0, stream>>>(hu, resid, en_g + e * Dm,
                                           en_b + e * Dm, G, out, e, e > 0, Mt);
  }
}
